// SpGraphTransAttentionLayer_44951127720194
// MI455X (gfx1250) — matrix operand order repaired and checked by reading, not compiled, not run on hardware
//
#include <hip/hip_runtime.h>
#include <hip/hip_bf16.h>

// Problem constants from the reference
#define B_  2
#define N_  20000
#define F_  256
#define A_  256
#define H_  8
#define DK_ 32
#define E_  320000
#define ROWS_ (B_ * N_)          // 40000 rows for the QKV GEMM

typedef __attribute__((ext_vector_type(16))) __bf16 bf16x16;
typedef __attribute__((ext_vector_type(8)))  __bf16 bf16x8;
typedef __attribute__((ext_vector_type(8)))  float  floatx8;
typedef __attribute__((ext_vector_type(4)))  int    v4i;

#define AS1 __attribute__((address_space(1)))
#define AS3 __attribute__((address_space(3)))

// CDNA5 async global->LDS copy path (ASYNCcnt-tracked), guarded so the file
// still compiles cleanly if this toolchain names the builtin differently.
#if __has_builtin(__builtin_amdgcn_global_load_async_to_lds_b128)
#define USE_ASYNC_LDS 1
#endif

#if __has_builtin(__builtin_amdgcn_s_wait_asynccnt)
#define WAIT_ASYNC0() __builtin_amdgcn_s_wait_asynccnt(0)
#else
#define WAIT_ASYNC0() asm volatile("s_wait_asynccnt 0x0" ::: "memory")
#endif

// ---------------------------------------------------------------------------
// Workspace layout (bytes):
//   [0)           : q   fp32, ROWS_*A_          = 40,960,000 B
//   [40,960,000)  : k   fp32, ROWS_*A_          = 40,960,000 B
//   [81,920,000)  : segmax u32, B_*N_*H_        =  1,280,000 B
//   [83,200,000)  : segsum f32, B_*N_*H_        =  1,280,000 B
//   [84,480,000)  : W frags bf16, 3*16*8*32*16  =    393,216 B
//   [84,873,216)  : x bf16, ROWS_*F_            = 20,480,000 B
// ---------------------------------------------------------------------------
#define WS_Q_OFF      0
#define WS_K_OFF      40960000
#define WS_SEGMAX_OFF 81920000
#define WS_SEGSUM_OFF 83200000
#define WS_WFRAG_OFF  84480000
#define WS_XBF_OFF    84873216

// d_out layout (floats): attention [0, B*E*H) | v_out [.., +B*N*DK*H) | prods
#define OUT_ATT_OFF   0
#define OUT_V_OFF     (B_ * E_ * H_)                    // 5,120,000
#define OUT_PRODS_OFF (OUT_V_OFF + B_ * N_ * DK_ * H_)  // 15,360,000

// ---------------------------------------------------------------------------
// 0) zero segmax + segsum (contiguous 640,000 u32)
// ---------------------------------------------------------------------------
__global__ void init_seg_kernel(unsigned int* __restrict__ seg) {
    int t = blockIdx.x * blockDim.x + threadIdx.x;
    if (t < 2 * B_ * N_ * H_) seg[t] = 0u;
}

// ---------------------------------------------------------------------------
// 1a) Convert x fp32 -> bf16 once (saves the 3x per-tile re-conversion and
//     lets the GEMM async-copy raw bf16 straight into LDS).
// ---------------------------------------------------------------------------
__global__ void xcvt_kernel(const float* __restrict__ x, __bf16* __restrict__ xb) {
    size_t t = (size_t)blockIdx.x * blockDim.x + threadIdx.x;   // over ROWS_*F_/4
    if (t >= (size_t)ROWS_ * F_ / 4) return;
    float4 f = ((const float4*)x)[t];
    __bf16* d = xb + t * 4;
    d[0] = (__bf16)f.x;  d[1] = (__bf16)f.y;
    d[2] = (__bf16)f.z;  d[3] = (__bf16)f.w;
}

// ---------------------------------------------------------------------------
// 1b) Pack Wq/Wk/Wv (fp32, A x F row-major) into bf16 B-matrix fragments in
//     the wave32 WMMA layout: element (k,n) -> lane = 16*((k%16)/8)+n, slot = 8*(k/16)+k%8.
//     Fragment = [mat][ntile*8 + kstep][lane][16 bf16]  (32 B per lane).
//     out[m,a] = sum_f x[m,f] * W[a,f]  =>  B[k=f][n=a] = W[a][f].
// ---------------------------------------------------------------------------
__global__ void pack_w_kernel(const float* __restrict__ Wq,
                              const float* __restrict__ Wk,
                              const float* __restrict__ Wv,
                              __bf16* __restrict__ wfrag) {
    int t = blockIdx.x * blockDim.x + threadIdx.x;   // 3*16*8*32 = 12288
    if (t >= 3 * 16 * 8 * 32) return;
    int lane = t & 31;  int r = t >> 5;
    int ks   = r & 7;   r >>= 3;
    int nt   = r & 15;  int mat = r >> 4;
    const float* W = (mat == 0) ? Wq : (mat == 1) ? Wk : Wv;
    int n     = nt * 16 + (lane & 15);
    int kbase = ks * 32 + 8 * (lane >> 4);
    __bf16* dst = wfrag + ((size_t)mat * 16 * 8 * 32 + ((nt * 8 + ks) * 32 + lane)) * 16;
#pragma unroll
    for (int i = 0; i < 16; ++i)
        dst[i] = (__bf16)W[(size_t)n * F_ + kbase + i + (i >= 8 ? 8 : 0)];
}

// ---------------------------------------------------------------------------
// 2) QKV GEMM: grid (2500, 3), block 128 (4 waves).
//    Each block: 16 rows x 256 channels of one matrix {q,k,v}.
//    bf16 x tile async-copied into LDS (ASYNCcnt) when the CDNA5 builtin is
//    available; A-fragments loaded per ISA 16-bit A layout (lane = 16g+m;
//    slots 0-7: K=8g+i, slots 8-15: K=16+8g+(i-8)).
//    8 k-steps x 4 n-tiles/wave -> 32 v_wmma_f32_16x16x32_bf16 per wave.
//    v is stored directly transposed to (B,N,DK,H).
// ---------------------------------------------------------------------------
__global__ void __launch_bounds__(128)
qkv_gemm_kernel(const __bf16* __restrict__ xbf,
                const __bf16* __restrict__ wfragAll,
                const float* __restrict__ bq,
                const float* __restrict__ bk,
                const float* __restrict__ bv,
                float* __restrict__ qout,
                float* __restrict__ kout,
                float* __restrict__ vout) {
    __shared__ __bf16 lx[16 * 256];   // 8 KB bf16 tile

    const int mt   = blockIdx.x;       // row tile
    const int mat  = blockIdx.y;       // 0=q 1=k 2=v
    const int row0 = mt * 16;
    const int tid  = threadIdx.x;

    const __bf16* xb = xbf + (size_t)row0 * F_;
#if defined(USE_ASYNC_LDS)
    // 512 x 16 B lane-transfers = 8 KB; 4 async b128 instructions per wave.
#pragma unroll
    for (int i = tid; i < 512; i += 128) {
        __builtin_amdgcn_global_load_async_to_lds_b128(
            (AS1 v4i*)(xb + (size_t)i * 8),
            (AS3 v4i*)(&lx[i * 8]), 0, 0);
    }
    WAIT_ASYNC0();
    __syncthreads();
#else
#pragma unroll
    for (int i = tid; i < 512; i += 128)
        ((int4*)lx)[i] = ((const int4*)xb)[i];
    __syncthreads();
#endif

    const int wave = tid >> 5;
    const int lane = tid & 31;
    const int g    = lane >> 4;     // K-half select
    const int ln   = lane & 15;     // M (A) / N (B,C,D)

    const float*  bias  = (mat == 0) ? bq : (mat == 1) ? bk : bv;
    const __bf16* wfrag = wfragAll + (size_t)mat * 16 * 8 * 32 * 16;

    // accumulators seeded with bias (C layout: N = ln, all 8 M-rows same N)
    floatx8 acc[4];
#pragma unroll
    for (int t = 0; t < 4; ++t) {
        float bv_ = bias[(wave * 4 + t) * 16 + ln];
#pragma unroll
        for (int r2 = 0; r2 < 8; ++r2) acc[t][r2] = bv_;
    }

#pragma unroll
    for (int ks = 0; ks < 8; ++ks) {
        const int k0 = ks * 32;
        // A fragment: row m=ln of LDS tile; two 16 B ds loads
        bf16x8 alo = *(const bf16x8*)(&lx[ln * 256 + k0 + 8 * g]);
        bf16x8 ahi = *(const bf16x8*)(&lx[ln * 256 + k0 + 16 + 8 * g]);
        bf16x16 a  = __builtin_shufflevector(alo, ahi,
                        0, 1, 2, 3, 4, 5, 6, 7, 8, 9, 10, 11, 12, 13, 14, 15);
#pragma unroll
        for (int t = 0; t < 4; ++t) {
            int nt = wave * 4 + t;
            bf16x16 bf = *(const bf16x16*)(wfrag + ((size_t)(nt * 8 + ks) * 32 + lane) * 16);
            acc[t] = __builtin_amdgcn_wmma_f32_16x16x32_bf16(
                false, a, false, bf, (short)0, acc[t], false, false);
        }
    }

    // store (D layout: M = r + 8g, N = n0 + ln)
    if (mat < 2) {
        float* dst = (mat == 0) ? qout : kout;
#pragma unroll
        for (int t = 0; t < 4; ++t) {
            int n0 = (wave * 4 + t) * 16;
#pragma unroll
            for (int r2 = 0; r2 < 8; ++r2)
                dst[(size_t)(row0 + r2 + 8 * g) * A_ + n0 + ln] = acc[t][r2];
        }
    } else {
        // v: write transposed (B,N,DK,H): channel a = h*32 + dk
#pragma unroll
        for (int t = 0; t < 4; ++t) {
            int a_ = (wave * 4 + t) * 16 + ln;
            int h  = a_ >> 5, dk = a_ & 31;
#pragma unroll
            for (int r2 = 0; r2 < 8; ++r2) {
                size_t gr = (size_t)(row0 + r2 + 8 * g);
                vout[(gr * DK_ + dk) * H_ + h] = acc[t][r2];
            }
        }
    }
}

// ---------------------------------------------------------------------------
// 3) Edge kernel: one wave per edge. lane -> head h = lane&7, chunk c = lane>>3
//    (dims c*8..c*8+7). Butterfly over lanes {h, h+8, h+16, h+24}.
//    prods > 0 always => segment-max via u32 atomicMax on the bit pattern.
// ---------------------------------------------------------------------------
__global__ void __launch_bounds__(256)
edge_kernel(const float* __restrict__ q, const float* __restrict__ k,
            const int* __restrict__ edge,
            const float* __restrict__ output_var,
            const float* __restrict__ lengthscale,
            float* __restrict__ prods, unsigned int* __restrict__ segmax) {
    int gw   = (blockIdx.x * blockDim.x + threadIdx.x) >> 5;  // edge over B*E
    int lane = threadIdx.x & 31;
    if (gw >= B_ * E_) return;
    int b = gw / E_, e = gw - b * E_;
    int src = edge[(size_t)b * 2 * E_ + e];
    int dst = edge[(size_t)b * 2 * E_ + E_ + e];

    int h = lane & 7, c = lane >> 3;
    const float4* q4 = (const float4*)(q + ((size_t)b * N_ + src) * A_ + h * 32 + c * 8);
    const float4* k4 = (const float4*)(k + ((size_t)b * N_ + dst) * A_ + h * 32 + c * 8);
    float4 qa = q4[0], qb = q4[1], ka = k4[0], kb = k4[1];
    float d0 = qa.x - ka.x, d1 = qa.y - ka.y, d2 = qa.z - ka.z, d3 = qa.w - ka.w;
    float d4 = qb.x - kb.x, d5 = qb.y - kb.y, d6 = qb.z - kb.z, d7 = qb.w - kb.w;
    float acc = d0*d0 + d1*d1 + d2*d2 + d3*d3 + d4*d4 + d5*d5 + d6*d6 + d7*d7;
    acc += __shfl_xor(acc, 8, 32);
    acc += __shfl_xor(acc, 16, 32);

    if (lane < 8) {  // lane == head
        float ov = output_var[0], ls = lengthscale[0];
        float p = ov * ov * __expf(-acc / (2.0f * ls * ls));
        prods[(size_t)gw * H_ + lane] = p;
        atomicMax(&segmax[((size_t)b * N_ + dst) * H_ + lane], __float_as_uint(p));
    }
}

// ---------------------------------------------------------------------------
// 4) exp-sum pass: pe = exp(p - segmax[dst]); atomicAdd into segsum.
// ---------------------------------------------------------------------------
__global__ void expsum_kernel(const float* __restrict__ prods,
                              const int* __restrict__ edge,
                              const unsigned int* __restrict__ segmax,
                              float* __restrict__ segsum) {
    size_t t = (size_t)blockIdx.x * blockDim.x + threadIdx.x;  // B*E*H
    if (t >= (size_t)B_ * E_ * H_) return;
    int h = (int)(t & 7);
    size_t be = t >> 3;
    int b = (int)(be / E_), e = (int)(be - (size_t)b * E_);
    int dst = edge[(size_t)b * 2 * E_ + E_ + e];
    float m  = __uint_as_float(segmax[((size_t)b * N_ + dst) * H_ + h]);
    atomicAdd(&segsum[((size_t)b * N_ + dst) * H_ + h], __expf(prods[t] - m));
}

// ---------------------------------------------------------------------------
// 5) normalize: attention = exp(p - m) / (s + 1e-16)
// ---------------------------------------------------------------------------
__global__ void norm_kernel(const float* __restrict__ prods,
                            const int* __restrict__ edge,
                            const unsigned int* __restrict__ segmax,
                            const float* __restrict__ segsum,
                            float* __restrict__ attention) {
    size_t t = (size_t)blockIdx.x * blockDim.x + threadIdx.x;
    if (t >= (size_t)B_ * E_ * H_) return;
    int h = (int)(t & 7);
    size_t be = t >> 3;
    int b = (int)(be / E_), e = (int)(be - (size_t)b * E_);
    int dst = edge[(size_t)b * 2 * E_ + E_ + e];
    size_t seg = ((size_t)b * N_ + dst) * H_ + h;
    float m = __uint_as_float(segmax[seg]);
    float s = segsum[seg];
    attention[t] = __expf(prods[t] - m) / (s + 1e-16f);
}

// ---------------------------------------------------------------------------
extern "C" void kernel_launch(void* const* d_in, const int* in_sizes, int n_in,
                              void* d_out, int out_size, void* d_ws, size_t ws_size,
                              hipStream_t stream) {
    const float* x   = (const float*)d_in[0];
    const float* Wq  = (const float*)d_in[1];
    const float* bq  = (const float*)d_in[2];
    const float* Wk  = (const float*)d_in[3];
    const float* bk  = (const float*)d_in[4];
    const float* Wv  = (const float*)d_in[5];
    const float* bv  = (const float*)d_in[6];
    const float* ov  = (const float*)d_in[7];
    const float* ls  = (const float*)d_in[8];
    const int*   edg = (const int*)d_in[9];

    float* out = (float*)d_out;
    float* attention = out + OUT_ATT_OFF;
    float* v_out     = out + OUT_V_OFF;
    float* prods     = out + OUT_PRODS_OFF;

    char* ws = (char*)d_ws;
    float*        qbuf   = (float*)(ws + WS_Q_OFF);
    float*        kbuf   = (float*)(ws + WS_K_OFF);
    unsigned int* segmax = (unsigned int*)(ws + WS_SEGMAX_OFF);
    float*        segsum = (float*)(ws + WS_SEGSUM_OFF);
    __bf16*       wfrag  = (__bf16*)(ws + WS_WFRAG_OFF);
    __bf16*       xbf    = (__bf16*)(ws + WS_XBF_OFF);

    // 0) zero segment max/sum buffers (contiguous: 2 * B*N*H u32)
    init_seg_kernel<<<(2 * B_ * N_ * H_ + 255) / 256, 256, 0, stream>>>(segmax);

    // 1) x fp32 -> bf16, and pre-swizzle weights into WMMA B-fragments
    xcvt_kernel<<<(ROWS_ * F_ / 4 + 255) / 256, 256, 0, stream>>>(x, xbf);
    pack_w_kernel<<<(3 * 16 * 8 * 32 + 255) / 256, 256, 0, stream>>>(Wq, Wk, Wv, wfrag);

    // 2) QKV GEMM (WMMA bf16, f32 accumulate; async LDS staging)
    qkv_gemm_kernel<<<dim3(ROWS_ / 16, 3), 128, 0, stream>>>(
        xbf, wfrag, bq, bk, bv, qbuf, kbuf, v_out);

    // 3) edge RBF scores + segment max
    {
        long long waves = (long long)B_ * E_;
        long long thr   = waves * 32;
        edge_kernel<<<(unsigned)((thr + 255) / 256), 256, 0, stream>>>(
            qbuf, kbuf, edg, ov, ls, prods, segmax);
    }

    // 4) segment exp-sum
    expsum_kernel<<<(B_ * E_ * H_ + 255) / 256, 256, 0, stream>>>(
        prods, edg, segmax, segsum);

    // 5) normalize -> attention
    norm_kernel<<<(B_ * E_ * H_ + 255) / 256, 256, 0, stream>>>(
        prods, edg, segmax, segsum, attention);
}
